// ISFOTransformer_13451837571934
// MI455X (gfx1250) — compile-verified, matrix operand order checked by reading, not run on hardware
//
#include <hip/hip_runtime.h>
#include <hip/hip_bf16.h>
#include <math.h>
#include <stdint.h>

// Problem constants (match reference)
#define BB 16
#define SS 512
#define IND 64
#define DM 64
#define HID 256
#define TY 256
#define TX 288
#define LF 33          // filter length
#define NT 5           // sample steps
#define RIDGE 0.01f
#define THRESH 0.005f

typedef __attribute__((ext_vector_type(2)))  float    v2f;
typedef __attribute__((ext_vector_type(8)))  float    v8f;
typedef __attribute__((ext_vector_type(16))) _Float16 v16h;
typedef __attribute__((ext_vector_type(4)))  uint32_t u32x4;
typedef __attribute__((ext_vector_type(8)))  uint32_t u32x8;

// ---------------------------------------------------------------------------
// Fragment loaders (CDNA5 WMMA VGPR layouts, ISA 7.12.2, wave32)
// ---------------------------------------------------------------------------

// f32 A-matrix 16x4 tile: lane l holds row m=l&15; lanes 0-15 K pair {k0,k0+1},
// lanes 16-31 K pair {k0+2,k0+3}.
__device__ __forceinline__ v2f frag_a_f32(const float* A, int lda, int k0, int lane) {
    int m  = lane & 15;
    int kb = k0 + ((lane >> 4) << 1);
    v2f a;
    a[0] = A[m * lda + kb];
    a[1] = A[m * lda + kb + 1];
    return a;
}

// f32 B-matrix 4x16 tile: lane l holds col n=l&15; VGPR v -> row K = kb+v.
__device__ __forceinline__ v2f frag_b_f32(const float* Bp, int ldb, int k0, int lane) {
    int n  = lane & 15;
    int kb = k0 + ((lane >> 4) << 1);
    v2f b;
    b[0] = Bp[kb * ldb + n];
    b[1] = Bp[(kb + 1) * ldb + n];
    return b;
}

// f16 A-matrix 16x32 tile loaded (with f32->f16 convert) from row-major f32.
__device__ __forceinline__ v16h frag_a_f16(const float* A, int lda, int k0, int lane) {
    int m  = lane & 15;
    int hi = lane >> 4;
    v16h a;
#pragma unroll
    for (int v = 0; v < 8; ++v) {
        int kb = k0 + ((v & 3) << 1) + ((v >> 2) << 4) + hi * 8;
        a[2 * v]     = (_Float16)A[m * lda + kb];
        a[2 * v + 1] = (_Float16)A[m * lda + kb + 1];
    }
    return a;
}

// f16 B-matrix 32x16 tile where B = K^T (so B[k][n] = Kmat[n][k]).
__device__ __forceinline__ v16h frag_bT_f16(const float* Kt, int ldk, int k0, int lane) {
    int n  = lane & 15;
    int hi = lane >> 4;
    v16h b;
#pragma unroll
    for (int v = 0; v < 8; ++v) {
        int kb = k0 + ((v & 3) << 1) + ((v >> 2) << 4) + hi * 8;
        b[2 * v]     = (_Float16)Kt[n * ldk + kb];
        b[2 * v + 1] = (_Float16)Kt[n * ldk + kb + 1];
    }
    return b;
}

__device__ __forceinline__ v8f wmma_f32(v2f a, v2f b, v8f c) {
    return __builtin_amdgcn_wmma_f32_16x16x4_f32(false, a, false, b, (short)0, c, false, false);
}
__device__ __forceinline__ v8f wmma_f16(v16h a, v16h b, v8f c) {
    return __builtin_amdgcn_wmma_f32_16x16x32_f16(false, a, false, b, (short)0, c, false, false);
}

// LDS byte offset of a generic pointer into __shared__ (flat LDS aperture maps
// addr[31:0] -> LDS address, ISA 10.2).
__device__ __forceinline__ uint32_t lds_off(const void* p) {
    return (uint32_t)(uintptr_t)p;
}

// ---------------------------------------------------------------------------
// Kernel 1: MLP  (h = LN(fps@W1+b1); e0 = tanh(tanh(h)@W2+b2);
//                 e = e0 + 0.05*PE;  q = e@Wq+bq; k = e@Wk+bk)
// One wave per 16 rows of the flattened (B*S, .) activation matrix.
// ---------------------------------------------------------------------------
__global__ __launch_bounds__(32)
void k1_mlp(const float* __restrict__ fps,
            const float* __restrict__ W1, const float* __restrict__ b1,
            const float* __restrict__ gamma, const float* __restrict__ beta,
            const float* __restrict__ W2, const float* __restrict__ b2,
            const float* __restrict__ Wq, const float* __restrict__ bq,
            const float* __restrict__ Wk, const float* __restrict__ bk,
            float* __restrict__ e_orig, float* __restrict__ qw, float* __restrict__ kw) {
    __shared__ float sh_h[16][HID + 4];   // hidden activations (stride 260)
    __shared__ float sh_e[16][DM];        // embedding with pos-enc

    const int  lane = threadIdx.x;
    const long r0   = (long)blockIdx.x * 16;       // flattened row base
    const float* A0 = fps + r0 * IND;

    // --- GEMM1: (16x64) @ (64x256) + b1 -> sh_h  (f32 WMMA)
    for (int nt = 0; nt < HID / 16; ++nt) {
        v8f c = {};
        for (int kk = 0; kk < IND; kk += 4) {
            v2f a = frag_a_f32(A0, IND, kk, lane);
            v2f b = frag_b_f32(W1 + nt * 16, HID, kk, lane);
            c = wmma_f32(a, b, c);
        }
        int n = lane & 15, hi = lane >> 4;
#pragma unroll
        for (int v = 0; v < 8; ++v)
            sh_h[v + 8 * hi][nt * 16 + n] = c[v] + b1[nt * 16 + n];
    }
    __syncthreads();

    // --- LayerNorm + tanh per row (wave-cooperative)
    for (int m = 0; m < 16; ++m) {
        float s = 0.f;
        for (int j = lane; j < HID; j += 32) s += sh_h[m][j];
        for (int o = 16; o > 0; o >>= 1) s += __shfl_xor(s, o, 32);
        float mu = s * (1.0f / HID);
        float vs = 0.f;
        for (int j = lane; j < HID; j += 32) { float d = sh_h[m][j] - mu; vs += d * d; }
        for (int o = 16; o > 0; o >>= 1) vs += __shfl_xor(vs, o, 32);
        float inv = rsqrtf(vs * (1.0f / HID) + 1e-5f);
        for (int j = lane; j < HID; j += 32) {
            float hn = (sh_h[m][j] - mu) * inv * gamma[j] + beta[j];
            sh_h[m][j] = tanhf(hn);
        }
    }
    __syncthreads();

    // --- GEMM2: (16x256) @ (256x64) + b2, tanh, +0.05*PE
    const float kPE = -0.14391156831212787f;       // -ln(10000)/64
    for (int nt = 0; nt < DM / 16; ++nt) {
        v8f c = {};
        for (int kk = 0; kk < HID; kk += 4) {
            v2f a = frag_a_f32(&sh_h[0][0], HID + 4, kk, lane);
            v2f b = frag_b_f32(W2 + nt * 16, DM, kk, lane);
            c = wmma_f32(a, b, c);
        }
        int n = lane & 15, hi = lane >> 4;
#pragma unroll
        for (int v = 0; v < 8; ++v) {
            int  m    = v + 8 * hi;
            long gr   = r0 + m;
            int  sidx = (int)(gr & (SS - 1));
            int  dcol = nt * 16 + n;
            float ev  = tanhf(c[v] + b2[dcol]);
            e_orig[gr * DM + dcol] = ev;
            float div = __expf((float)(dcol & ~1) * kPE);
            float ang = (float)sidx * div;
            float pe  = (dcol & 1) ? __cosf(ang) : __sinf(ang);
            sh_e[m][dcol] = ev + 0.05f * pe;
        }
    }
    __syncthreads();

    // --- GEMM3: q = e@Wq+bq ; k = e@Wk+bk   (16x64 @ 64x64 each)
    for (int nt = 0; nt < DM / 16; ++nt) {
        v8f cq = {}, ck = {};
        for (int kk = 0; kk < DM; kk += 4) {
            v2f a   = frag_a_f32(&sh_e[0][0], DM, kk, lane);
            v2f bq_ = frag_b_f32(Wq + nt * 16, DM, kk, lane);
            v2f bk_ = frag_b_f32(Wk + nt * 16, DM, kk, lane);
            cq = wmma_f32(a, bq_, cq);
            ck = wmma_f32(a, bk_, ck);
        }
        int n = lane & 15, hi = lane >> 4;
#pragma unroll
        for (int v = 0; v < 8; ++v) {
            long gr   = r0 + v + 8 * hi;
            int  dcol = nt * 16 + n;
            qw[gr * DM + dcol] = cq[v] + bq[dcol];
            kw[gr * DM + dcol] = ck[v] + bk[dcol];
        }
    }
}

// ---------------------------------------------------------------------------
// Kernel 2: causal attention  alpha = softmax(2 * (q k^T)/sqrt(d), causal)
// One wave computes a 16-row x 512-col score stripe with f16 WMMA.
// ---------------------------------------------------------------------------
__global__ __launch_bounds__(32)
void k2_attn(const float* __restrict__ qw, const float* __restrict__ kw,
             float* __restrict__ alpha) {
    __shared__ float sc[16][SS];
    const int lane = threadIdx.x;
    const int b    = blockIdx.x >> 5;     // 32 row tiles per batch
    const int rt   = blockIdx.x & 31;
    const float* qb = qw + ((long)b * SS + rt * 16) * DM;
    const float* kb = kw + (long)b * SS * DM;

    // q-tile fragments are reused across all 32 column tiles
    v16h a0 = frag_a_f16(qb, DM, 0, lane);
    v16h a1 = frag_a_f16(qb, DM, 32, lane);

    for (int ct = 0; ct < SS / 16; ++ct) {
        v16h b0 = frag_bT_f16(kb + ct * 16 * DM, DM, 0, lane);
        v16h b1 = frag_bT_f16(kb + ct * 16 * DM, DM, 32, lane);
        v8f c = {};
        c = wmma_f16(a0, b0, c);
        c = wmma_f16(a1, b1, c);
        int n = lane & 15, hi = lane >> 4;
#pragma unroll
        for (int v = 0; v < 8; ++v) sc[v + 8 * hi][ct * 16 + n] = c[v];
    }
    __syncthreads();

    // causal mask + softmax over 512 columns, per row
    for (int m = 0; m < 16; ++m) {
        int   row = rt * 16 + m;
        float lg[16];
        float mx = -3.0e38f;
#pragma unroll
        for (int i = 0; i < 16; ++i) {
            int j = lane + 32 * i;
            float lv = (j <= row) ? sc[m][j] * 0.25f : -2.0e9f;   // 2/sqrt(64)=0.25
            lg[i] = lv;
            mx = fmaxf(mx, lv);
        }
        for (int o = 16; o > 0; o >>= 1) mx = fmaxf(mx, __shfl_xor(mx, o, 32));
        float sum = 0.f;
#pragma unroll
        for (int i = 0; i < 16; ++i) { lg[i] = __expf(lg[i] - mx); sum += lg[i]; }
        for (int o = 16; o > 0; o >>= 1) sum += __shfl_xor(sum, o, 32);
        float inv = 1.0f / sum;
        long base = ((long)b * SS + row) * SS;
#pragma unroll
        for (int i = 0; i < 16; ++i) alpha[base + lane + 32 * i] = lg[i] * inv;
    }
}

// ---------------------------------------------------------------------------
// Kernel 3: per-(b,s) Gram matrices.  X[t,l] = x[b,s,t+l] (256x33).
// G = X^T X (33x33), C = X^T y.  x row staged into LDS by the Tensor Data
// Mover (tensor_load_to_lds, 1-row 288-elem tile), y row staged by
// global_load_async_to_lds_b32. One fragment serves as both A(X^T) and B(X).
// ---------------------------------------------------------------------------
__global__ __launch_bounds__(32)
void k3_gram(const float* __restrict__ x, const float* __restrict__ y,
             float* __restrict__ G, float* __restrict__ Cv) {
    __shared__ float xs[TX];
    __shared__ float ys[TY];
    const int  lane = threadIdx.x;
    const long bs   = blockIdx.x;
    const float* xr = x + bs * TX;
    const float* yr = y + bs * TY;

    // --- TDM: stage x row (288 f32) into LDS with one tensor DMA op.
    {
        uint64_t ga    = (uint64_t)(uintptr_t)xr;
        uint32_t lbase = lds_off(&xs[0]);
        u32x4 g0;
        g0[0] = 1u;                                            // count=1, user mode
        g0[1] = lbase;                                         // lds_addr
        g0[2] = (uint32_t)ga;                                  // global_addr[31:0]
        g0[3] = (uint32_t)((ga >> 32) & 0x01FFFFFFull)         // global_addr[56:32]
              | (2u << 30);                                    // type=2 (image)
        u32x8 g1;
        g1[0] = (2u << 16);          // data_size=2 (4 bytes), no multicast/pad
        g1[1] = ((uint32_t)TX) << 16;// tensor_dim0[15:0] @ bits 63:48
        g1[2] = (1u << 16);          // tensor_dim0 hi=0; tensor_dim1=1 @ bits 95:80
        g1[3] = ((uint32_t)TX) << 16;// tensor_dim1 hi=0; tile_dim0=288 @ bits 127:112
        g1[4] = 0u;                  // tile_dim1=0 (unused), tile_dim2=0
        g1[5] = (uint32_t)TX;        // tensor_dim0_stride low 32
        g1[6] = 0u;
        g1[7] = 0u;
        asm volatile("tensor_load_to_lds %0, %1, null, null"
                     :: "s"(g0), "s"(g1)
                     : "memory");
    }

    // --- Async copy: stage y row (256 f32) global -> LDS, 1 dword per lane.
    for (int j = lane; j < TY; j += 32) {
        uint32_t ldst = lds_off(&ys[j]);
        uint32_t goff = (uint32_t)(j * 4);
        asm volatile("global_load_async_to_lds_b32 %0, %1, %2"
                     :: "v"(ldst), "v"(goff), "s"(yr)
                     : "memory");
    }
    asm volatile("s_wait_asynccnt 0x0" ::: "memory");
    __builtin_amdgcn_s_wait_tensorcnt((short)0);
    __syncthreads();

    v8f z = {};
    v8f acc[3][3];
#pragma unroll
    for (int i = 0; i < 3; ++i)
#pragma unroll
        for (int j = 0; j < 3; ++j) acc[i][j] = z;

    const int m  = lane & 15;
    const int hi = lane >> 4;
    for (int kk = 0; kk < TY; kk += 4) {
        int kb = kk + hi * 2;
        v2f f[3];
#pragma unroll
        for (int t = 0; t < 3; ++t) {
            int  l  = t * 16 + m;       // row (for A) == col (for B) in tile t
            bool ok = l < LF;
            f[t][0] = ok ? xs[kb + l] : 0.f;
            f[t][1] = ok ? xs[kb + 1 + l] : 0.f;
        }
#pragma unroll
        for (int i = 0; i < 3; ++i)
#pragma unroll
            for (int j = 0; j < 3; ++j) acc[i][j] = wmma_f32(f[i], f[j], acc[i][j]);
    }

    float* Gp = G + bs * (LF * LF);
#pragma unroll
    for (int i = 0; i < 3; ++i)
#pragma unroll
        for (int j = 0; j < 3; ++j)
#pragma unroll
            for (int v = 0; v < 8; ++v) {
                int r  = i * 16 + v + 8 * hi;
                int cc = j * 16 + (lane & 15);
                if (r < LF && cc < LF) Gp[r * LF + cc] = acc[i][j][v];
            }

    float* Cp = Cv + bs * LF;
    for (int l = lane; l < LF; l += 32) {
        float s = 0.f;
        for (int t = 0; t < TY; ++t) s += xs[t + l] * ys[t];
        Cp[l] = s;
    }
}

// ---------------------------------------------------------------------------
// Kernel 4: thresholded attention weights -> gram/cross accumulation.
// One block per batch b: G[b] is streamed ONCE and accumulated into all NT
// weighted sums (5x less HBM traffic than per-(b,t) blocks). Zero-weight s
// (sparse after causal softmax + threshold) skipped uniformly.
// ---------------------------------------------------------------------------
__global__ __launch_bounds__(256)
void k4_build(const float* __restrict__ alpha, const int* __restrict__ steps,
              const float* __restrict__ G, const float* __restrict__ Cv,
              float* __restrict__ gram, float* __restrict__ cross,
              float* __restrict__ validf) {
    __shared__ float wm[NT][SS];
    __shared__ int   anyv[NT];
    const int b   = blockIdx.x;
    const int tid = threadIdx.x;
    if (tid < NT) anyv[tid] = 0;
    __syncthreads();

    for (int t = 0; t < NT; ++t) {
        const int st = steps[t];
        const float* arow = alpha + ((long)b * SS + st) * SS;
        for (int s = tid; s < SS; s += 256) {
            float w  = arow[s];
            float wv = (w > THRESH) ? w : 0.f;
            wm[t][s] = wv;
            if (wv > 0.f) anyv[t] = 1;    // benign racy flag set
        }
    }
    __syncthreads();

    float acc[NT][5] = {};
    const float* Gb = G + (long)b * SS * (LF * LF);
    for (int s = 0; s < SS; ++s) {
        float wv[NT];
        float wsum = 0.f;
#pragma unroll
        for (int t = 0; t < NT; ++t) { wv[t] = wm[t][s]; wsum += wv[t]; }
        if (wsum == 0.f) continue;                   // uniform skip
        const float* Gs = Gb + (long)s * (LF * LF);
        if (s + 1 < SS) __builtin_prefetch(Gb + (long)(s + 1) * (LF * LF), 0, 1);
        float g[5];
#pragma unroll
        for (int u = 0; u < 5; ++u) {
            int e = u * 256 + tid;
            g[u] = (e < LF * LF) ? Gs[e] : 0.f;
        }
#pragma unroll
        for (int t = 0; t < NT; ++t)
#pragma unroll
            for (int u = 0; u < 5; ++u) acc[t][u] += wv[t] * g[u];
    }
    for (int t = 0; t < NT; ++t) {
        float* gp = gram + ((long)b * NT + t) * (LF * LF);
#pragma unroll
        for (int u = 0; u < 5; ++u) {
            int e = u * 256 + tid;
            if (e < LF * LF) {
                int l = e / LF, mcol = e % LF;
                gp[e] = acc[t][u] + ((l == mcol) ? RIDGE : 0.f);
            }
        }
    }
    if (tid < NT * LF) {
        const int t = tid / LF, l = tid % LF;
        const float* Cb = Cv + (long)b * SS * LF;
        float s2 = 0.f;
        for (int s = 0; s < SS; ++s) {
            float wv = wm[t][s];
            if (wv != 0.f) s2 += wv * Cb[s * LF + l];
        }
        cross[((long)b * NT + t) * LF + l] = s2;
    }
    __syncthreads();
    if (tid < NT) validf[b * NT + tid] = anyv[tid] ? 1.f : 0.f;
}

// ---------------------------------------------------------------------------
// Kernel 5: solve gram * w = cross (33x33, SPD + ridge -> no pivoting),
// then prediction MSE at the sampled step. One wave per (b,t).
// ---------------------------------------------------------------------------
__global__ __launch_bounds__(32)
void k5_solve(const float* __restrict__ gram, const float* __restrict__ cross,
              const float* __restrict__ x, const float* __restrict__ y,
              const int* __restrict__ steps, const float* __restrict__ validf,
              float* __restrict__ loss_bt) {
    __shared__ float M[LF][LF + 2];   // augmented: col 33 = rhs
    __shared__ float wt[LF];
    const int lane = threadIdx.x;
    const int b    = blockIdx.x / NT;
    const int t    = blockIdx.x % NT;
    const float* gp = gram + (long)blockIdx.x * (LF * LF);
    const float* cp = cross + (long)blockIdx.x * LF;
    for (int e = lane; e < LF * LF; e += 32) M[e / LF][e % LF] = gp[e];
    for (int l = lane; l < LF; l += 32) M[l][LF] = cp[l];
    __syncthreads();

    for (int c = 0; c < LF; ++c) {
        float inv = 1.0f / M[c][c];
        __syncthreads();
        for (int r = c + 1 + lane; r < LF; r += 32) {
            float f = M[r][c] * inv;
            for (int cc = c + 1; cc <= LF; ++cc) M[r][cc] -= f * M[c][cc];
        }
        __syncthreads();
    }
    if (lane == 0) {
        for (int r = LF - 1; r >= 0; --r) {
            float s = M[r][LF];
            for (int cc = r + 1; cc < LF; ++cc) s -= M[r][cc] * wt[cc];
            wt[r] = s / M[r][r];
        }
    }
    __syncthreads();

    const int st = steps[t];
    const float* xr = x + ((long)b * SS + st) * TX;
    const float* yr = y + ((long)b * SS + st) * TY;
    float se = 0.f;
    for (int tk = lane; tk < TY; tk += 32) {
        float p = 0.f;
#pragma unroll
        for (int l = 0; l < LF; ++l) p += xr[tk + l] * wt[l];
        float d = yr[tk] - p;
        se += d * d;
    }
    for (int o = 16; o > 0; o >>= 1) se += __shfl_xor(se, o, 32);
    if (lane == 0)
        loss_bt[blockIdx.x] = (se * (1.0f / TY)) * validf[blockIdx.x];
}

// ---------------------------------------------------------------------------
// Kernel 6: deterministic fixed-order reduction of the 80 (b,t) losses.
// ---------------------------------------------------------------------------
__global__ __launch_bounds__(128)
void k6_reduce(const float* __restrict__ loss_bt, const float* __restrict__ validf,
               float* __restrict__ out_loss) {
    __shared__ float sl[128];
    __shared__ float sv[128];
    const int tid = threadIdx.x;
    sl[tid] = (tid < BB * NT) ? loss_bt[tid] : 0.f;
    sv[tid] = (tid < BB * NT) ? validf[tid] : 0.f;
    __syncthreads();
    for (int o = 64; o > 0; o >>= 1) {
        if (tid < o) { sl[tid] += sl[tid + o]; sv[tid] += sv[tid + o]; }
        __syncthreads();
    }
    if (tid == 0) {
        float n = (sv[0] < 1.f) ? 1.f : sv[0];
        out_loss[0] = sl[0] / n;
    }
}

// ---------------------------------------------------------------------------
// Launch
// ---------------------------------------------------------------------------
extern "C" void kernel_launch(void* const* d_in, const int* in_sizes, int n_in,
                              void* d_out, int out_size, void* d_ws, size_t ws_size,
                              hipStream_t stream) {
    const float* fps   = (const float*)d_in[0];
    const float* x     = (const float*)d_in[1];
    const float* y     = (const float*)d_in[2];
    const int*   steps = (const int*)d_in[3];
    const float* W1    = (const float*)d_in[4];
    const float* b1    = (const float*)d_in[5];
    const float* gamma = (const float*)d_in[6];
    const float* beta  = (const float*)d_in[7];
    const float* W2    = (const float*)d_in[8];
    const float* b2    = (const float*)d_in[9];
    const float* Wq    = (const float*)d_in[10];
    const float* bq    = (const float*)d_in[11];
    const float* Wk    = (const float*)d_in[12];
    const float* bk    = (const float*)d_in[13];

    float* out    = (float*)d_out;
    float* loss   = out;                                   // [1]
    float* alpha  = out + 1;                               // [B*S*S]
    float* e_orig = out + 1 + (long)BB * SS * SS;          // [B*S*D]

    float* ws = (float*)d_ws;
    size_t o = 0;
    float* qw     = ws + o; o += (size_t)BB * SS * DM;         // 0.52M
    float* kw     = ws + o; o += (size_t)BB * SS * DM;         // 0.52M
    float* G      = ws + o; o += (size_t)BB * SS * LF * LF;    // 8.9M
    float* Cv     = ws + o; o += (size_t)BB * SS * LF;         // 0.27M
    float* gram   = ws + o; o += (size_t)BB * NT * LF * LF;
    float* cross  = ws + o; o += (size_t)BB * NT * LF;
    float* lossbt = ws + o; o += (size_t)BB * NT;
    float* validf = ws + o; o += (size_t)BB * NT;
    (void)ws_size; (void)in_sizes; (void)n_in; (void)out_size;

    k1_mlp  <<<(BB * SS) / 16, 32, 0, stream>>>(fps, W1, b1, gamma, beta, W2, b2,
                                                Wq, bq, Wk, bk, e_orig, qw, kw);
    k2_attn <<<BB * (SS / 16), 32, 0, stream>>>(qw, kw, alpha);
    k3_gram <<<BB * SS, 32, 0, stream>>>(x, y, G, Cv);
    k4_build<<<BB, 256, 0, stream>>>(alpha, steps, G, Cv, gram, cross, validf);
    k5_solve<<<BB * NT, 32, 0, stream>>>(gram, cross, x, y, steps, validf, lossbt);
    k6_reduce<<<1, 128, 0, stream>>>(lossbt, validf, loss);
}
